// remodel_7705171329682
// MI455X (gfx1250) — matrix operand order repaired and checked by reading, not compiled, not run on hardware
//
#include <hip/hip_runtime.h>
#include <hip/hip_bf16.h>

typedef __attribute__((ext_vector_type(16))) __bf16 v16bf;
typedef __attribute__((ext_vector_type(8)))  __bf16 v8bf;
typedef __attribute__((ext_vector_type(8)))  float  v8f;

#define E_    100
#define H_    230
#define T_    70
#define BIG_  128
#define N_    4096
#define NR_   53
#define PD_   5
#define IN_   110    // E + 2*PD
#define G3_   690    // 3*H
#define GP_   720    // gate dim padded to 45 tiles of 16
#define KI_   128    // IN padded
#define KH_   256    // H padded
#define HPAD_ 240    // per-gate padded out dim (15 tiles of 16)
#define XST_  136    // proj LDS bf16 row stride (68 words == 4 mod 64)
#define HST_  264    // gru  LDS bf16 row stride (132 words == 4 mod 64)

// ---------------- weight prep: f32 -> padded bf16 ----------------
__global__ void prep_kernel(const float* __restrict__ src, __bf16* __restrict__ dst,
                            int srcRows, int srcCols, int dstRows, int dstCols,
                            int gateRegroup) {
    int idx = blockIdx.x * blockDim.x + threadIdx.x;
    if (idx >= dstRows * dstCols) return;
    int rr = idx / dstCols, c = idx % dstCols;
    int g = -1;
    if (gateRegroup) {
        int gate = rr / HPAD_, i = rr % HPAD_;
        if (i < H_) g = gate * H_ + i;
    } else if (rr < srcRows) {
        g = rr;
    }
    float v = (g >= 0 && c < srcCols) ? src[g * srcCols + c] : 0.f;
    dst[idx] = (__bf16)v;
}

// Build one A fragment (16x32 bf16) from bf16 LDS, contiguous K order:
// elements 0..7  <-> K = kbase + 16*half + 0..7
// elements 8..15 <-> K = kbase + 16*half + 8..15
__device__ __forceinline__ v16bf load_afrag(const __bf16* row, int kbase, int half) {
    v8bf lo = *(const v8bf*)(row + kbase + 16 * half);
    v8bf hi = *(const v8bf*)(row + kbase + 16 * half + 8);
    return __builtin_shufflevector(lo, hi, 0, 1, 2, 3, 4, 5, 6, 7,
                                   8, 9, 10, 11, 12, 13, 14, 15);
}

// ---------------- embedding gather + input projection GEMM ----------------
// block = 16 rows of flattened (t,n); 8 waves split 45 gate tiles.
__global__ void __launch_bounds__(256) proj_kernel(
        const int* __restrict__ sentence, const int* __restrict__ pos1,
        const int* __restrict__ pos2,
        const float* __restrict__ wemb, const float* __restrict__ p1emb,
        const float* __restrict__ p2emb,
        const __bf16* __restrict__ wih, const float* __restrict__ bih,
        __bf16* __restrict__ xp) {
    __shared__ __bf16 x_bf[16 * XST_];
    const int tid  = threadIdx.x;
    const int lane = tid & 31, wave = tid >> 5;
    const int ln   = lane & 15, half = lane >> 4;
    const int rbase = blockIdx.x * 16;
    const int t  = rbase / N_;
    const int nb = rbase % N_;

    for (int idx = tid; idx < 16 * XST_; idx += 256) {
        int m = idx / XST_, i = idx % XST_;
        int n = nb + m;
        float v = 0.f;
        if (i < E_) {
            int sid = sentence[n * T_ + t];
            v = wemb[(size_t)sid * E_ + i];
        } else if (i < E_ + PD_) {
            int p = pos1[n * T_ + t];
            v = p1emb[p * PD_ + (i - E_)];
        } else if (i < IN_) {
            int p = pos2[n * T_ + t];
            v = p2emb[p * PD_ + (i - E_ - PD_)];
        }
        x_bf[m * XST_ + i] = (__bf16)v;
    }
    __syncthreads();

    v16bf a[4];
#pragma unroll
    for (int kk = 0; kk < 4; ++kk)
        a[kk] = load_afrag(x_bf + ln * XST_, kk * 32, half);

    for (int j = wave; j < 45; j += 8) {
        // preload all B fragments, then run the WMMA chain (single wait)
        v16bf b[4];
#pragma unroll
        for (int kk = 0; kk < 4; ++kk)
            b[kk] = *(const v16bf*)(wih + (size_t)(j * 16 + ln) * KI_ + kk * 32 + 16 * half);
        v8f acc = {};
#pragma unroll
        for (int kk = 0; kk < 4; ++kk)
            acc = __builtin_amdgcn_wmma_f32_16x16x32_bf16(
                false, a[kk], false, b[kk], (short)0, acc, false, false);
        int g = j * 16 + ln;
        if (g < G3_) {
            float bias = bih[g];
#pragma unroll
            for (int r = 0; r < 8; ++r) {
                int m = r + 8 * half;
                xp[((size_t)t * N_ + nb + m) * G3_ + g] = (__bf16)(acc[r] + bias);
            }
        }
    }
}

// ---------------- GRU scan ----------------
// block = 512 threads (16 waves), owns 16 sentences. Wave w (<15) owns output
// tile w and keeps its 24 w_hh B-fragments (3 gates x 8 K-steps) in registers
// for the whole 70-step scan. h kept in LDS twice: bf16 (A-fragment source,
// two ds_load_b128 per fragment) and f32 (exact state for z*h_old).
__global__ void __launch_bounds__(512) gru_kernel(
        const __bf16* __restrict__ xp, const __bf16* __restrict__ whh,
        const float* __restrict__ bhh, float* __restrict__ hout, int reverse) {
    __shared__ __bf16 h_bf[16 * HST_];
    __shared__ float  h_f[16 * 257];
    const int tid  = threadIdx.x;
    const int lane = tid & 31, wave = tid >> 5;
    const int ln   = lane & 15, half = lane >> 4;
    const int nbase = blockIdx.x * 16;
    const bool active = wave < 15;
    const int off = wave * 16;
    const int jg  = off + ln;
    const bool valid = active && (jg < H_);

    for (int i = tid; i < 16 * (HST_ / 2); i += 512) ((unsigned*)h_bf)[i] = 0u;
    for (int i = tid; i < 16 * 257; i += 512) h_f[i] = 0.f;

    // hoist recurrent weights into registers (L2 -> VGPR, once)
    v16bf Bf[3][8];
    if (active) {
#pragma unroll
        for (int g = 0; g < 3; ++g)
#pragma unroll
            for (int kk = 0; kk < 8; ++kk)
                Bf[g][kk] = *(const v16bf*)(whh + (size_t)(g * HPAD_ + off + ln) * KH_ +
                                            kk * 32 + 16 * half);
    }
    float br_ = 0.f, bz_ = 0.f, bn_ = 0.f;
    if (valid) { br_ = bhh[jg]; bz_ = bhh[H_ + jg]; bn_ = bhh[2 * H_ + jg]; }
    __syncthreads();

    for (int s = 0; s < T_; ++s) {
        const int t = reverse ? (T_ - 1 - s) : s;
        v16bf a[8];
        if (active) {
#pragma unroll
            for (int kk = 0; kk < 8; ++kk)
                a[kk] = load_afrag(h_bf + ln * HST_, kk * 32, half);
        }
        __syncthreads();

        if (active) {
            v8f accr = {}, accz = {}, accn = {};
#pragma unroll
            for (int kk = 0; kk < 8; ++kk) {
                accr = __builtin_amdgcn_wmma_f32_16x16x32_bf16(false, a[kk], false, Bf[0][kk], (short)0, accr, false, false);
                accz = __builtin_amdgcn_wmma_f32_16x16x32_bf16(false, a[kk], false, Bf[1][kk], (short)0, accz, false, false);
                accn = __builtin_amdgcn_wmma_f32_16x16x32_bf16(false, a[kk], false, Bf[2][kk], (short)0, accn, false, false);
            }
            if (valid) {
                const int tn = reverse ? (t - 1) : (t + 1);
#pragma unroll
                for (int r = 0; r < 8; ++r) {
                    int m = r + 8 * half;
                    int nrow = nbase + m;
                    size_t xb = ((size_t)t * N_ + nrow) * (size_t)G3_ + jg;
                    float xr = (float)xp[xb];
                    float xz = (float)xp[xb + H_];
                    float xn = (float)xp[xb + 2 * H_];
                    if (s + 1 < T_) {  // prefetch next timestep's gates
                        size_t xbn = ((size_t)tn * N_ + nrow) * (size_t)G3_ + jg;
                        __builtin_prefetch((const void*)(xp + xbn), 0, 1);
                        __builtin_prefetch((const void*)(xp + xbn + 2 * H_), 0, 1);
                    }
                    float rg = 1.f / (1.f + __expf(-(xr + accr[r] + br_)));
                    float zg = 1.f / (1.f + __expf(-(xz + accz[r] + bz_)));
                    float ng = tanhf(xn + rg * (accn[r] + bn_));
                    float hold = h_f[m * 257 + jg];
                    float hnew = (1.f - zg) * ng + zg * hold;
                    h_f[m * 257 + jg] = hnew;
                    h_bf[m * HST_ + jg] = (__bf16)hnew;
                    hout[((size_t)nrow * T_ + t) * H_ + jg] = hnew;
                }
            }
        }
        __syncthreads();
    }
}

// ---------------- word-level attention per sentence ----------------
__global__ void __launch_bounds__(256) wordatt_kernel(
        const float* __restrict__ hf, const float* __restrict__ hb,
        const float* __restrict__ attw, float* __restrict__ senrep) {
    __shared__ float sc[T_];
    __shared__ float sden;
    const int n = blockIdx.x;
    const int tid = threadIdx.x, lane = tid & 31, wave = tid >> 5;
    const float* hfp = hf + (size_t)n * T_ * H_;
    const float* hbp = hb + (size_t)n * T_ * H_;

    for (int t = wave; t < T_; t += 8) {
        float s = 0.f;
        for (int h = lane; h < H_; h += 32)
            s += tanhf(hfp[t * H_ + h] + hbp[t * H_ + h]) * attw[h];
#pragma unroll
        for (int o = 16; o > 0; o >>= 1) s += __shfl_down(s, o);
        if (lane == 0) sc[t] = s;
    }
    __syncthreads();
    if (tid == 0) {
        float m = -1e30f;
        for (int t = 0; t < T_; ++t) m = fmaxf(m, sc[t]);
        float d = 0.f;
        for (int t = 0; t < T_; ++t) { float e = __expf(sc[t] - m); sc[t] = e; d += e; }
        sden = d;
    }
    __syncthreads();
    if (tid < H_) {
        float acc = 0.f;
        for (int t = 0; t < T_; ++t)
            acc += sc[t] * (hfp[t * H_ + tid] + hbp[t * H_ + tid]);
        senrep[(size_t)n * H_ + tid] = tanhf(acc / sden);
    }
}

// ---------------- bag attention + logits + prob/acc/loss ----------------
__global__ void __launch_bounds__(256) bag_kernel(
        const float* __restrict__ senrep, const int* __restrict__ tshape,
        const float* __restrict__ sena, const float* __restrict__ senr,
        const float* __restrict__ relemb, const float* __restrict__ send,
        const float* __restrict__ ybatch,
        float* __restrict__ out_acc, float* __restrict__ out_prob,
        float* __restrict__ bag_loss) {
    __shared__ float red[256];
    __shared__ float sen_s[H_];
    __shared__ float logits[NR_];
    __shared__ float sdenom;
    const int b = blockIdx.x, tid = threadIdx.x;
    const int start = tshape[b], end = tshape[b + 1];

    float lmax = -1e30f;
    for (int s = start + tid; s < end; s += 256) {
        const float* sp = senrep + (size_t)s * H_;
        float e = 0.f;
        for (int h = 0; h < H_; ++h) e += sp[h] * sena[h] * senr[h];
        lmax = fmaxf(lmax, e);
    }
    red[tid] = lmax;
    __syncthreads();
    for (int o = 128; o > 0; o >>= 1) {
        if (tid < o) red[tid] = fmaxf(red[tid], red[tid + o]);
        __syncthreads();
    }
    const float m = red[0];
    if (tid < H_) sen_s[tid] = 0.f;
    if (tid == 0) sdenom = 0.f;
    __syncthreads();

    for (int s = start + tid; s < end; s += 256) {
        const float* sp = senrep + (size_t)s * H_;
        float e = 0.f;
        for (int h = 0; h < H_; ++h) e += sp[h] * sena[h] * senr[h];
        float ex = __expf(e - m);
        atomicAdd(&sdenom, ex);
        for (int h = 0; h < H_; ++h) atomicAdd(&sen_s[h], ex * sp[h]);
    }
    __syncthreads();

    if (tid < NR_) {
        float inv = 1.f / sdenom;
        float lg = 0.f;
        for (int h = 0; h < H_; ++h) lg += sen_s[h] * inv * relemb[tid * H_ + h];
        logits[tid] = lg + send[tid];
    }
    __syncthreads();

    if (tid == 0) {
        float mm = -1e30f; int pred = 0;
        for (int r = 0; r < NR_; ++r)
            if (logits[r] > mm) { mm = logits[r]; pred = r; }
        float d = 0.f;
        for (int r = 0; r < NR_; ++r) d += __expf(logits[r] - mm);
        float loss = 0.f, ymax = -1e30f; int ytrue = 0;
        for (int r = 0; r < NR_; ++r) {
            float lg = logits[r];
            out_prob[b * NR_ + r] = __expf(lg - mm) / d;
            float y = ybatch[b * NR_ + r];
            float sp_pos = fmaxf(lg, 0.f) + log1pf(__expf(-fabsf(lg)));   // softplus(lg)
            float sp_neg = fmaxf(-lg, 0.f) + log1pf(__expf(-fabsf(lg)));  // softplus(-lg)
            loss += y * sp_neg + (1.f - y) * sp_pos;
            if (y > ymax) { ymax = y; ytrue = r; }
        }
        bag_loss[b] = loss / (float)NR_;
        out_acc[b] = (pred == ytrue) ? 1.f : 0.f;
    }
}

__global__ void lossred_kernel(const float* __restrict__ bag_loss, float* __restrict__ out) {
    __shared__ float red[BIG_];
    int tid = threadIdx.x;
    red[tid] = bag_loss[tid];
    __syncthreads();
    for (int o = BIG_ / 2; o > 0; o >>= 1) {
        if (tid < o) red[tid] += red[tid + o];
        __syncthreads();
    }
    if (tid == 0) out[0] = red[0];
}

extern "C" void kernel_launch(void* const* d_in, const int* in_sizes, int n_in,
                              void* d_out, int out_size, void* d_ws, size_t ws_size,
                              hipStream_t stream) {
    const int*   sentence = (const int*)d_in[0];
    const int*   pos1     = (const int*)d_in[1];
    const int*   pos2     = (const int*)d_in[2];
    const int*   tshape   = (const int*)d_in[3];
    const float* ybatch   = (const float*)d_in[4];
    const float* wemb     = (const float*)d_in[5];
    const float* p1emb    = (const float*)d_in[6];
    const float* p2emb    = (const float*)d_in[7];
    const float* wihf     = (const float*)d_in[8];
    const float* whhf     = (const float*)d_in[9];
    const float* bihf     = (const float*)d_in[10];
    const float* bhhf     = (const float*)d_in[11];
    const float* wihb     = (const float*)d_in[12];
    const float* whhb     = (const float*)d_in[13];
    const float* bihb     = (const float*)d_in[14];
    const float* bhhb     = (const float*)d_in[15];
    const float* attw     = (const float*)d_in[16];
    const float* sena     = (const float*)d_in[17];
    const float* senr     = (const float*)d_in[18];
    const float* relemb   = (const float*)d_in[19];
    const float* send     = (const float*)d_in[20];

    char* ws = (char*)d_ws;
    size_t off = 0;
    auto alloc = [&](size_t bytes) -> void* {
        void* p = ws + off;
        off = (off + bytes + 511) & ~(size_t)511;
        return p;
    };
    __bf16* WihF = (__bf16*)alloc((size_t)GP_ * KI_ * 2);
    __bf16* WihB = (__bf16*)alloc((size_t)GP_ * KI_ * 2);
    __bf16* WhhF = (__bf16*)alloc((size_t)GP_ * KH_ * 2);
    __bf16* WhhB = (__bf16*)alloc((size_t)GP_ * KH_ * 2);
    __bf16* XpF  = (__bf16*)alloc((size_t)T_ * N_ * G3_ * 2);
    __bf16* XpB  = (__bf16*)alloc((size_t)T_ * N_ * G3_ * 2);
    float*  Hf   = (float*)alloc((size_t)N_ * T_ * H_ * 4);
    float*  Hb   = (float*)alloc((size_t)N_ * T_ * H_ * 4);
    float*  Srep = (float*)alloc((size_t)N_ * H_ * 4);
    float*  BagL = (float*)alloc((size_t)BIG_ * 4);

    float* outf     = (float*)d_out;
    float* out_loss = outf;            // [1]
    float* out_acc  = outf + 1;        // [128]
    float* out_prob = outf + 1 + BIG_; // [128*53]

    // weight prep
    prep_kernel<<<(GP_ * KI_ + 255) / 256, 256, 0, stream>>>(wihf, WihF, G3_, IN_, GP_, KI_, 0);
    prep_kernel<<<(GP_ * KI_ + 255) / 256, 256, 0, stream>>>(wihb, WihB, G3_, IN_, GP_, KI_, 0);
    prep_kernel<<<(GP_ * KH_ + 255) / 256, 256, 0, stream>>>(whhf, WhhF, G3_, H_, GP_, KH_, 1);
    prep_kernel<<<(GP_ * KH_ + 255) / 256, 256, 0, stream>>>(whhb, WhhB, G3_, H_, GP_, KH_, 1);

    // input projections (embedding gather fused)
    const int projBlocks = (N_ * T_) / 16;  // 17920
    proj_kernel<<<projBlocks, 256, 0, stream>>>(sentence, pos1, pos2, wemb, p1emb, p2emb,
                                                WihF, bihf, XpF);
    proj_kernel<<<projBlocks, 256, 0, stream>>>(sentence, pos1, pos2, wemb, p1emb, p2emb,
                                                WihB, bihb, XpB);

    // GRU scans (forward / backward)
    gru_kernel<<<N_ / 16, 512, 0, stream>>>(XpF, WhhF, bhhf, Hf, 0);
    gru_kernel<<<N_ / 16, 512, 0, stream>>>(XpB, WhhB, bhhb, Hb, 1);

    // word attention -> sentence representation
    wordatt_kernel<<<N_, 256, 0, stream>>>(Hf, Hb, attw, Srep);

    // bag attention + logits + prob/acc/per-bag loss
    bag_kernel<<<BIG_, 256, 0, stream>>>(Srep, tshape, sena, senr, relemb, send, ybatch,
                                         out_acc, out_prob, BagL);

    // total loss
    lossred_kernel<<<1, BIG_, 0, stream>>>(BagL, out_loss);
}
